// DMASA_4509715661328
// MI455X (gfx1250) — matrix operand order repaired and checked by reading, not compiled, not run on hardware
//
#include <hip/hip_runtime.h>
#include <hip/hip_bf16.h>
#include <math.h>

// ---- problem constants ----
#define B_      16
#define L_      4096
#define N_      64
#define D_      64
#define H_      4
#define WV_     16
#define LAYERS_ 2
#define TOPK_   16
#define T_      256      // L / WAVE
#define DFF_    256
#define DH_     16
#define BT_     (B_*T_)          // 4096
#define ROWS_   (BT_*N_)         // 262144
#define KBIG_   (T_*D_)          // 16384

typedef __attribute__((ext_vector_type(16))) __bf16 v16bf;
typedef __attribute__((ext_vector_type(8)))  float  v8f;

__device__ inline v8f zero8() {
  v8f z;
#pragma unroll
  for (int i = 0; i < 8; ++i) z[i] = 0.0f;
  return z;
}

__device__ inline v8f wmma_bf16(v16bf a, v16bf b, v8f c) {
  return __builtin_amdgcn_wmma_f32_16x16x32_bf16(false, a, false, b, (short)0, c, false, false);
}

// A-fragment layout: position of k (0..31) inside a 32-wide K block; identity,
// so lane half h reads k = 16h..16h+15 (same contiguous order as B).
__device__ inline int aswz(int k) {
  return k;
}

// A fragment from A-swizzled storage; per lane: 32 contiguous bytes.
__device__ inline v16bf frag_a_swz(const __bf16* s, int ld) {
  int lane = threadIdx.x & 31, m = lane & 15, half = lane >> 4;
  return *(const v16bf*)(s + m*ld + half*16);
}
// B fragment from B-swizzled storage (((k/32)*Nn + n)*32 + k%32); 32B per lane.
__device__ inline v16bf frag_b_swz(const __bf16* s) {
  int lane = threadIdx.x & 31, n = lane & 15, half = lane >> 4;
  return *(const v16bf*)(s + n*32 + half*16);
}
// C/D 16x16 f32: lane = half*16+n ; VGPR r holds row m = r + 8*half
__device__ inline void dcoord(int& r0, int& n) {
  int lane = threadIdx.x & 31;
  n  = lane & 15;
  r0 = (lane >> 4) << 3;
}

// ===================== 1) cosine top-k graph bias =====================
__global__ void graph_bias_kernel(const float* __restrict__ emb, float* __restrict__ bias) {
  __shared__ float nrm[N_];
  int i = threadIdx.x;          // 64 threads, one row each
  float s = 0.0f;
  for (int d = 0; d < D_; ++d) { float v = emb[i*D_ + d]; s += v*v; }
  nrm[i] = sqrtf(s);
  __syncthreads();
  float cosv[N_];
  for (int j = 0; j < N_; ++j) {
    float dot = 0.0f;
    for (int d = 0; d < D_; ++d) dot += emb[i*D_ + d] * emb[j*D_ + d];
    cosv[j] = dot / (nrm[i] * nrm[j]);
  }
  bool sel[N_];
  for (int j = 0; j < N_; ++j) sel[j] = false;
  for (int k = 0; k < TOPK_; ++k) {
    int best = 0; float bv = -1e30f;
    for (int j = 0; j < N_; ++j)
      if (!sel[j] && cosv[j] > bv) { bv = cosv[j]; best = j; }
    sel[best] = true;
  }
  for (int j = 0; j < N_; ++j) bias[i*N_ + j] = sel[j] ? 0.0f : -1e9f;
}

// ========== 2) f32 -> bf16 weight conversion WITH B-fragment swizzle ==========
// dst[((k/32)*Nn + n)*32 + (k%32)] = bf16(src[k*Nn + n]); grid = (K/32, Nn/64)
__global__ void __launch_bounds__(256) cvt_swz_kernel(
    const float* __restrict__ src, __bf16* __restrict__ dst, int Nn) {
  int kt = blockIdx.x, n0 = blockIdx.y * 64;
  __shared__ __bf16 tile[32][64];
  for (int e = threadIdx.x; e < 32*64; e += 256) {
    int r = e >> 6, c = e & 63;
    tile[r][c] = (__bf16)src[((size_t)kt*32 + r)*Nn + n0 + c];
  }
  __syncthreads();
  for (int e = threadIdx.x; e < 32*64; e += 256) {
    int n = e >> 5, kloc = e & 31;
    dst[((size_t)kt*Nn + n0 + n)*32 + kloc] = tile[kloc][n];
  }
}

// ===================== 3) grouped strided conv (+ reference's transposition) ====
// h[bt, n, d] = sum_w x[b, t*16+w, d] * conv_w[d, n, w] + conv_b[d, n]
__global__ void conv_kernel(const float* __restrict__ x, const float* __restrict__ cw,
                            const float* __restrict__ cb, float* __restrict__ h) {
  int bt = blockIdx.x;
  int b = bt >> 8, t = bt & 255;
  __shared__ float xs[WV_][D_];
  for (int e = threadIdx.x; e < WV_*D_; e += 256) {
    int w = e >> 6, d = e & 63;
    xs[w][d] = x[((size_t)b*L_ + t*WV_ + w)*N_ + d];
  }
  __syncthreads();
  for (int e = threadIdx.x; e < N_*D_; e += 256) {
    int n = e >> 6, d = e & 63;
    float acc = cb[d*D_ + n];
    const float* cwp = &cw[((size_t)d*D_ + n)*WV_];
#pragma unroll
    for (int w = 0; w < WV_; ++w) acc += xs[w][d] * cwp[w];
    h[((size_t)bt*N_ + n)*D_ + d] = acc;
  }
}

// ===================== 4) hin = bf16(h + emb), vectorized =====================
__global__ void addemb_kernel(const float* __restrict__ h, const float* __restrict__ emb,
                              __bf16* __restrict__ hin, int nchunk /* total/8 */) {
  int i = blockIdx.x * blockDim.x + threadIdx.x;
  int stride = gridDim.x * blockDim.x;
  for (; i < nchunk; i += stride) {
    int off8 = i << 3;
    int eoff = off8 & (N_*D_ - 1);
    float4 h0 = *(const float4*)(h + off8);
    float4 h1 = *(const float4*)(h + off8 + 4);
    float4 e0 = *(const float4*)(emb + eoff);
    float4 e1 = *(const float4*)(emb + eoff + 4);
    __bf16 tmp[8] = {(__bf16)(h0.x+e0.x), (__bf16)(h0.y+e0.y), (__bf16)(h0.z+e0.z), (__bf16)(h0.w+e0.w),
                     (__bf16)(h1.x+e1.x), (__bf16)(h1.y+e1.y), (__bf16)(h1.z+e1.z), (__bf16)(h1.w+e1.w)};
    *(uint4*)(hin + off8) = *(const uint4*)tmp;
  }
}

// ===================== 5) fused QKV projection (per (b,t) token) =====================
__global__ void __launch_bounds__(256) qkv_kernel(
    const __bf16* __restrict__ hin,
    const __bf16* __restrict__ wq, const __bf16* __restrict__ wk, const __bf16* __restrict__ wv,
    const float* __restrict__ bq, const float* __restrict__ bk, const float* __restrict__ bv,
    __bf16* __restrict__ q, __bf16* __restrict__ k, __bf16* __restrict__ v) {
  int bt = blockIdx.x;
  __shared__ __align__(32) __bf16 As[N_][D_];   // A-swizzled input
  __shared__ __align__(16) __bf16 xo[N_][D_];   // output repack tile
  const uint4* srcv = (const uint4*)(hin + (size_t)bt*N_*D_);
  for (int e = threadIdx.x; e < 512; e += 256) {
    int r = e >> 3, k0 = (e & 7) << 3;
    int kd = (k0 & 32) | aswz(k0 & 31);
    *(uint4*)&As[r][kd] = srcv[e];
  }
  __syncthreads();
  int wavei = threadIdx.x >> 5;
  int laneN, r0; dcoord(r0, laneN);
  const __bf16* wm[3] = {wq, wk, wv};
  const float*  bm[3] = {bq, bk, bv};
  __bf16*       om[3] = {q, k, v};
  for (int mat = 0; mat < 3; ++mat) {
#pragma unroll
    for (int t2 = 0; t2 < 2; ++t2) {
      int tile = wavei*2 + t2;            // 16 tiles over 8 waves
      int mt = tile >> 2, nt = tile & 3;
      v8f acc = zero8();
#pragma unroll
      for (int kk = 0; kk < 64; kk += 32) {
        v16bf a = frag_a_swz(&As[mt*16][kk], D_);
        v16bf b = frag_b_swz(wm[mat] + ((kk >> 5)*D_ + nt*16)*32);
        acc = wmma_bf16(a, b, acc);
      }
      int col = nt*16 + laneN;
      float bias_v = bm[mat][col];
#pragma unroll
      for (int i = 0; i < 8; ++i)
        xo[mt*16 + r0 + i][col] = (__bf16)(acc[i] + bias_v);
    }
    __syncthreads();
    uint4* dst = (uint4*)(om[mat] + (size_t)bt*N_*D_);
    const uint4* s = (const uint4*)&xo[0][0];
    for (int e = threadIdx.x; e < 512; e += 256) dst[e] = s[e];
    __syncthreads();
  }
}

// ===================== 6) fused masked attention (scores/softmax/PV) =====================
// DH=16 == one K-half of a 32-K fragment: load full head-pair K=32 fragments and
// zero the B fragment in lanes whose K-half != current head (kills cross-head terms).
__global__ void __launch_bounds__(256) attn_kernel(
    const __bf16* __restrict__ q, const __bf16* __restrict__ k,
    const __bf16* __restrict__ v, const float* __restrict__ bias,
    __bf16* __restrict__ o /* stored A-swizzled for owo */) {
  int bt = blockIdx.x;
  __shared__ __align__(32) __bf16 qs[N_][32];       // 4 KB: A-swz q slice; reused as o repack
  __shared__ __align__(32) __bf16 ks[N_][32];       // 4 KB: B-swz k slice
  __shared__ __align__(32) __bf16 vs[2*N_*32];      // 8 KB: B-swz v
  __shared__ float sc[2][N_][N_];                   // 32 KB
  __shared__ __align__(32) __bf16 ab[2][N_][N_];    // 16 KB: A-swz probs
  int wavei = threadIdx.x >> 5;
  int laneN, r0; dcoord(r0, laneN);
  int khalf = (threadIdx.x >> 4) & 1;
  const size_t base = (size_t)bt*N_*D_;
  // stage v (B-swizzled) once
  {
    const uint4* vv = (const uint4*)(v + base);
    for (int e = threadIdx.x; e < 512; e += 256) {
      int node = e >> 3, d0 = (e & 7) << 3;
      uint4 pk = vv[e];
      __bf16 tmp[8];
      *(uint4*)tmp = pk;
#pragma unroll
      for (int u = 0; u < 8; ++u)
        vs[(((node >> 5)*64) + d0 + u)*32 + (node & 31)] = tmp[u];
    }
  }
  for (int hp = 0; hp < 2; ++hp) {     // two heads per pass
    // stage q (A-swz) / k (B-swz) columns [hp*32, hp*32+32) -- pure vector copies
    {
      int e = threadIdx.x;
      int r = e >> 2, c0 = (e & 3) << 3;
      *(uint4*)&qs[r][aswz(c0)] = *(const uint4*)(q + base + (size_t)r*D_ + hp*32 + c0);
      *(uint4*)&ks[r][c0]       = *(const uint4*)(k + base + (size_t)r*D_ + hp*32 + c0);
    }
    __syncthreads();
    // scores = q_h @ k_h^T * 0.25 + bias
    for (int tile = wavei; tile < 32; tile += 8) {
      int hl = tile >> 4, rem = tile & 15, mt = rem >> 2, nt = rem & 3;
      v16bf a = frag_a_swz(&qs[mt*16][0], 32);
      v16bf b = frag_b_swz(&ks[nt*16][0]);
      if (khalf != hl) {
#pragma unroll
        for (int u = 0; u < 16; ++u) b[u] = (__bf16)0.0f;
      }
      v8f c = wmma_bf16(a, b, zero8());
#pragma unroll
      for (int i = 0; i < 8; ++i) {
        int row = mt*16 + r0 + i, col = nt*16 + laneN;
        sc[hl][row][col] = c[i]*0.25f + bias[row*N_ + col];
      }
    }
    __syncthreads();
    // row softmax (2 heads x 64 rows) -> A-swizzled bf16 probs
    if (threadIdx.x < 128) {
      int hl = threadIdx.x >> 6, row = threadIdx.x & 63;
      float m = -1e30f;
      for (int j = 0; j < N_; ++j) m = fmaxf(m, sc[hl][row][j]);
      float s = 0.0f;
      for (int j = 0; j < N_; ++j) {
        float e = __expf(sc[hl][row][j] - m);
        sc[hl][row][j] = e; s += e;
      }
      float inv = 1.0f / s;
      for (int j0 = 0; j0 < N_; j0 += 8) {
        __bf16 tmp[8];
#pragma unroll
        for (int u = 0; u < 8; ++u) tmp[u] = (__bf16)(sc[hl][row][j0+u] * inv);
        int kd = (j0 & 32) | aswz(j0 & 31);
        *(uint4*)&ab[hl][row][kd] = *(const uint4*)tmp;
      }
    }
    __syncthreads();
    // o_h = attn_h @ v_h; write D-frags into qs (free now), then coalesced store
    for (int tile = wavei; tile < 8; tile += 8) {
      int hl = tile >> 2, mt = tile & 3;
      int head = hp*2 + hl;
      v8f c = zero8();
#pragma unroll
      for (int kk = 0; kk < 64; kk += 32) {
        v16bf a = frag_a_swz(&ab[hl][mt*16][kk], N_);
        v16bf b = frag_b_swz(&vs[(((kk >> 5)*64) + head*DH_)*32]);
        c = wmma_bf16(a, b, c);
      }
      int pos = aswz(hl*DH_ + laneN);   // A-swz position within this 32-block
#pragma unroll
      for (int i = 0; i < 8; ++i)
        qs[mt*16 + r0 + i][pos] = (__bf16)(c[i]);
    }
    __syncthreads();
    {
      int e = threadIdx.x;
      int r = e >> 2, c0 = (e & 3) << 3;
      *(uint4*)(o + base + (size_t)r*D_ + hp*32 + c0) = *(const uint4*)&qs[r][c0];
    }
    __syncthreads();
  }
}

// ===================== 7) O-proj + residual + LayerNorm (fused) =====================
__global__ void __launch_bounds__(256) owo_ln_kernel(
    const __bf16* __restrict__ o /* A-swizzled */, const __bf16* __restrict__ wo,
    const float* __restrict__ bo, const float* __restrict__ g, const float* __restrict__ be,
    const float* __restrict__ res, int layer, float* __restrict__ h) {
  int bt = blockIdx.x;
  __shared__ float xo[N_][D_];
  int wavei = threadIdx.x >> 5;
  int laneN, r0; dcoord(r0, laneN);
  const size_t base = (size_t)bt*N_*D_;
  for (int tile = wavei; tile < 16; tile += 8) {
    int mt = tile >> 2, nt = tile & 3;
    v8f c = zero8();
#pragma unroll
    for (int kk = 0; kk < 64; kk += 32) {
      v16bf a = frag_a_swz(o + base + (size_t)(mt*16)*D_ + kk, D_);
      v16bf b = frag_b_swz(wo + ((kk >> 5)*D_ + nt*16)*32);
      c = wmma_bf16(a, b, c);
    }
#pragma unroll
    for (int i = 0; i < 8; ++i) xo[mt*16 + r0 + i][nt*16 + laneN] = c[i];
  }
  __syncthreads();
  if (threadIdx.x < N_) {
    int row = threadIdx.x;
    float r1 = res[layer];
    float y[D_];
    float mean = 0.0f;
#pragma unroll
    for (int d = 0; d < D_; ++d) {
      y[d] = h[base + row*D_ + d] + r1*(xo[row][d] + bo[d]);
      mean += y[d];
    }
    mean *= (1.0f / D_);
    float var = 0.0f;
#pragma unroll
    for (int d = 0; d < D_; ++d) { float t = y[d] - mean; var += t*t; }
    var *= (1.0f / D_);
    float rinv = rsqrtf(var + 1e-5f);
#pragma unroll
    for (int d = 0; d < D_; ++d)
      h[base + row*D_ + d] = (y[d] - mean)*rinv*g[d] + be[d];
  }
}

// ===================== 8) fused FFN (GEMM+GELU+GEMM) + residual + LN =====================
__global__ void __launch_bounds__(256) ffn_ln_kernel(
    const __bf16* __restrict__ w1, const __bf16* __restrict__ w2,
    const float* __restrict__ b1, const float* __restrict__ b2,
    const float* __restrict__ g, const float* __restrict__ be,
    const float* __restrict__ res, int layer, float* __restrict__ h) {
  int bt = blockIdx.x;
  __shared__ __align__(32) __bf16 As[N_][D_];      // 8 KB, A-swizzled
  __shared__ __align__(32) __bf16 t1[N_][DFF_];    // 32 KB, A-swizzled per 32-block
  __shared__ float  xo[N_][D_];                    // 16 KB
  const size_t base = (size_t)bt*N_*D_;
  for (int e = threadIdx.x; e < 512; e += 256) {
    int r = e >> 3, k0 = (e & 7) << 3;
    float4 f0 = *(const float4*)&h[base + (size_t)r*D_ + k0];
    float4 f1 = *(const float4*)&h[base + (size_t)r*D_ + k0 + 4];
    __bf16 tmp[8] = {(__bf16)f0.x, (__bf16)f0.y, (__bf16)f0.z, (__bf16)f0.w,
                     (__bf16)f1.x, (__bf16)f1.y, (__bf16)f1.z, (__bf16)f1.w};
    int kd = (k0 & 32) | aswz(k0 & 31);
    *(uint4*)&As[r][kd] = *(const uint4*)tmp;
  }
  __syncthreads();
  int wavei = threadIdx.x >> 5;
  int laneN, r0; dcoord(r0, laneN);
  // t1 = gelu(h @ W1 + b1)   (M=64, N=256, K=64)
  for (int tile = wavei; tile < 64; tile += 8) {
    int mt = tile >> 4, nt = tile & 15;
    v8f c = zero8();
#pragma unroll
    for (int kk = 0; kk < 64; kk += 32) {
      v16bf a = frag_a_swz(&As[mt*16][kk], D_);
      v16bf b = frag_b_swz(w1 + ((size_t)(kk >> 5)*DFF_ + nt*16)*32);
      c = wmma_bf16(a, b, c);
    }
    int col = nt*16 + laneN;
    int coff = (col & ~31) | aswz(col & 31);
    float bb = b1[col];
#pragma unroll
    for (int i = 0; i < 8; ++i) {
      float xv = c[i] + bb;
      float t  = 0.7978845608028654f * (xv + 0.044715f*xv*xv*xv);
      t1[mt*16 + r0 + i][coff] = (__bf16)(0.5f*xv*(1.0f + tanhf(t)));
    }
  }
  __syncthreads();
  // f = t1 @ W2   (M=64, N=64, K=256)
  for (int tile = wavei; tile < 16; tile += 8) {
    int mt = tile >> 2, nt = tile & 3;
    v8f c = zero8();
#pragma unroll
    for (int kk = 0; kk < DFF_; kk += 32) {
      v16bf a = frag_a_swz(&t1[mt*16][kk], DFF_);
      v16bf b = frag_b_swz(w2 + ((size_t)(kk >> 5)*D_ + nt*16)*32);
      c = wmma_bf16(a, b, c);
    }
#pragma unroll
    for (int i = 0; i < 8; ++i) xo[mt*16 + r0 + i][nt*16 + laneN] = c[i];
  }
  __syncthreads();
  if (threadIdx.x < N_) {
    int row = threadIdx.x;
    float r2 = res[layer];
    float y[D_];
    float mean = 0.0f;
#pragma unroll
    for (int d = 0; d < D_; ++d) {
      y[d] = h[base + row*D_ + d] + r2*(xo[row][d] + b2[d]);
      mean += y[d];
    }
    mean *= (1.0f / D_);
    float var = 0.0f;
#pragma unroll
    for (int d = 0; d < D_; ++d) { float t = y[d] - mean; var += t*t; }
    var *= (1.0f / D_);
    float rinv = rsqrtf(var + 1e-5f);
#pragma unroll
    for (int d = 0; d < D_; ++d)
      h[base + row*D_ + d] = (y[d] - mean)*rinv*g[d] + be[d];
  }
}

// ===================== 9) readout GEMM: M=1024(B*N), N=4096(L), K=16384 =====================
__device__ inline void readout_stage(const float* __restrict__ h, __bf16 (*As)[32],
                                     int bmI, int kt) {
  int e = threadIdx.x;               // exactly one 8-run per thread
  int r = e >> 2, k0 = (e & 3) << 3;
  int grow = bmI*64 + r;
  int b = grow >> 6, n = grow & 63;
  int kidx = kt*32 + k0;
  int t = kidx >> 6, d = kidx & 63;
  const float* src = &h[(((size_t)b*T_ + t)*N_ + n)*D_ + d];
  float4 f0 = *(const float4*)src;
  float4 f1 = *(const float4*)(src + 4);
  __bf16 tmp[8] = {(__bf16)f0.x, (__bf16)f0.y, (__bf16)f0.z, (__bf16)f0.w,
                   (__bf16)f1.x, (__bf16)f1.y, (__bf16)f1.z, (__bf16)f1.w};
  *(uint4*)&As[r][aswz(k0)] = *(const uint4*)tmp;
}

__global__ void __launch_bounds__(256) readout_kernel(
    const float* __restrict__ h, const __bf16* __restrict__ lw /* B-swizzled */,
    const float* __restrict__ lb, float* __restrict__ out) {
  int bmI = blockIdx.x;    // 16 row tiles of 64 (one batch b per block)
  int bnI = blockIdx.y;    // 64 col tiles of 64
  __shared__ __align__(32) __bf16 As[2][64][32];   // 8 KB, A-swizzled, double-buffered
  __shared__ float xo[64][64];                     // 16 KB output repack
  int wavei = threadIdx.x >> 5;
  int laneN, r0; dcoord(r0, laneN);
  v8f acc0 = zero8(), acc1 = zero8();
  int mt0 = (wavei*2) >> 2,      nt0 = (wavei*2) & 3;
  int mt1 = (wavei*2 + 1) >> 2,  nt1 = (wavei*2 + 1) & 3;
  readout_stage(h, As[0], bmI, 0);
  __syncthreads();
  for (int kt = 0; kt < KBIG_/32; ++kt) {
    if (kt + 1 < KBIG_/32) {
      readout_stage(h, As[(kt + 1) & 1], bmI, kt + 1);   // stage next while computing
      __builtin_prefetch(lw + ((size_t)(kt+1)*L_ + bnI*64)*32 + (threadIdx.x << 4), 0, 1);
    }
    const __bf16 (*A)[32] = As[kt & 1];
    {
      v16bf a = frag_a_swz(&A[mt0*16][0], 32);
      v16bf b = frag_b_swz(lw + ((size_t)kt*L_ + bnI*64 + nt0*16)*32);
      acc0 = wmma_bf16(a, b, acc0);
    }
    {
      v16bf a = frag_a_swz(&A[mt1*16][0], 32);
      v16bf b = frag_b_swz(lw + ((size_t)kt*L_ + bnI*64 + nt1*16)*32);
      acc1 = wmma_bf16(a, b, acc1);
    }
    __syncthreads();
  }
  // epilogue: repack through LDS as [l][node] then coalesced float4 stores
#pragma unroll
  for (int t2 = 0; t2 < 2; ++t2) {
    int mt = t2 ? mt1 : mt0, nt = t2 ? nt1 : nt0;
    float lbv = lb[bnI*64 + nt*16 + laneN];
    v8f acc = t2 ? acc1 : acc0;
#pragma unroll
    for (int i = 0; i < 8; ++i)
      xo[nt*16 + laneN][mt*16 + r0 + i] = acc[i] + lbv;
  }
  __syncthreads();
  {
    int b = bmI, l0 = bnI*64;        // rows of this block are (b = bmI, node 0..63)
    for (int e = threadIdx.x; e < 1024; e += 256) {
      int lr = e >> 4, c4 = (e & 15) << 2;
      *(float4*)(out + ((size_t)b*L_ + l0 + lr)*N_ + c4) = *(const float4*)&xo[lr][c4];
    }
  }
}

// ===================== host side =====================
extern "C" void kernel_launch(void* const* d_in, const int* in_sizes, int n_in,
                              void* d_out, int out_size, void* d_ws, size_t ws_size,
                              hipStream_t stream) {
  const float* x    = (const float*)d_in[0];
  const float* emb  = (const float*)d_in[1];
  const float* cw   = (const float*)d_in[2];
  const float* cb   = (const float*)d_in[3];
  const float* Wq   = (const float*)d_in[4];
  const float* bq   = (const float*)d_in[5];
  const float* Wk   = (const float*)d_in[6];
  const float* bk   = (const float*)d_in[7];
  const float* Wv   = (const float*)d_in[8];
  const float* bv   = (const float*)d_in[9];
  const float* Wo   = (const float*)d_in[10];
  const float* bo   = (const float*)d_in[11];
  const float* ln1g = (const float*)d_in[12];
  const float* ln1b = (const float*)d_in[13];
  const float* ln2g = (const float*)d_in[14];
  const float* ln2b = (const float*)d_in[15];
  const float* W1   = (const float*)d_in[16];
  const float* b1   = (const float*)d_in[17];
  const float* W2   = (const float*)d_in[18];
  const float* b2   = (const float*)d_in[19];
  const float* res1 = (const float*)d_in[20];
  const float* res2 = (const float*)d_in[21];
  const float* lw   = (const float*)d_in[22];
  const float* lb   = (const float*)d_in[23];
  float* out = (float*)d_out;
  (void)in_sizes; (void)n_in; (void)out_size; (void)ws_size;

  char* wsb = (char*)d_ws;
  size_t off = 0;
  auto alloc = [&](size_t bytes) -> char* {
    char* p = wsb + off;
    off += (bytes + 255) & ~(size_t)255;
    return p;
  };
  float*  bias = (float*)alloc((size_t)N_*N_*4);
  __bf16* wqb  = (__bf16*)alloc((size_t)LAYERS_*D_*D_*2);
  __bf16* wkb  = (__bf16*)alloc((size_t)LAYERS_*D_*D_*2);
  __bf16* wvb  = (__bf16*)alloc((size_t)LAYERS_*D_*D_*2);
  __bf16* wob  = (__bf16*)alloc((size_t)LAYERS_*D_*D_*2);
  __bf16* w1b  = (__bf16*)alloc((size_t)LAYERS_*D_*DFF_*2);
  __bf16* w2b  = (__bf16*)alloc((size_t)LAYERS_*DFF_*D_*2);
  __bf16* lwb  = (__bf16*)alloc((size_t)KBIG_*L_*2);
  float*  h    = (float*)alloc((size_t)ROWS_*D_*4);
  __bf16* hinb = (__bf16*)alloc((size_t)ROWS_*D_*2);
  __bf16* qb   = (__bf16*)alloc((size_t)ROWS_*D_*2);
  __bf16* kb   = (__bf16*)alloc((size_t)ROWS_*D_*2);
  __bf16* vb   = (__bf16*)alloc((size_t)ROWS_*D_*2);
  __bf16* ob   = (__bf16*)alloc((size_t)ROWS_*D_*2);

  graph_bias_kernel<<<1, 64, 0, stream>>>(emb, bias);
  // swizzling bf16 conversion (stacked-layer weights treated as tall K x N)
  cvt_swz_kernel<<<dim3(LAYERS_*D_/32, 1),       256, 0, stream>>>(Wq, wqb, D_);
  cvt_swz_kernel<<<dim3(LAYERS_*D_/32, 1),       256, 0, stream>>>(Wk, wkb, D_);
  cvt_swz_kernel<<<dim3(LAYERS_*D_/32, 1),       256, 0, stream>>>(Wv, wvb, D_);
  cvt_swz_kernel<<<dim3(LAYERS_*D_/32, 1),       256, 0, stream>>>(Wo, wob, D_);
  cvt_swz_kernel<<<dim3(LAYERS_*D_/32, DFF_/64), 256, 0, stream>>>(W1, w1b, DFF_);
  cvt_swz_kernel<<<dim3(LAYERS_*DFF_/32, 1),     256, 0, stream>>>(W2, w2b, D_);
  cvt_swz_kernel<<<dim3(KBIG_/32, L_/64),        256, 0, stream>>>(lw, lwb, L_);
  conv_kernel<<<BT_, 256, 0, stream>>>(x, cw, cb, h);

  for (int l = 0; l < LAYERS_; ++l) {
    addemb_kernel<<<2048, 256, 0, stream>>>(h, emb, hinb, ROWS_*D_/8);
    qkv_kernel<<<BT_, 256, 0, stream>>>(hinb,
        wqb + (size_t)l*D_*D_, wkb + (size_t)l*D_*D_, wvb + (size_t)l*D_*D_,
        bq + l*D_, bk + l*D_, bv + l*D_, qb, kb, vb);
    attn_kernel<<<BT_, 256, 0, stream>>>(qb, kb, vb, bias, ob);
    owo_ln_kernel<<<BT_, 256, 0, stream>>>(ob, wob + (size_t)l*D_*D_, bo + l*D_,
        ln1g + l*D_, ln1b + l*D_, res1, l, h);
    ffn_ln_kernel<<<BT_, 256, 0, stream>>>(w1b + (size_t)l*D_*DFF_, w2b + (size_t)l*DFF_*D_,
        b1 + l*DFF_, b2 + l*D_, ln2g + l*D_, ln2b + l*D_, res2, l, h);
  }

  dim3 rg(16, 64);
  readout_kernel<<<rg, 256, 0, stream>>>(h, lwb, lb, out);
}
